// BalancedKmeans_6614249636299
// MI455X (gfx1250) — compile-verified, matrix operand order checked by reading, not run on hardware
//
#include <hip/hip_runtime.h>

// ---------------------------------------------------------------------------
// Balanced K-means for MI455X (gfx1250, wave32, WMMA, async-to-LDS).
//   dist GEMM  : B tile staged in LDS via global_load_async_to_lds_b128,
//                v_wmma_f32_16x16x32_bf16, fp32 norms added in epilogue
//   argsort    : per-row 256-key bitonic sort in LDS (key = d2 bits | idx)
//   assignment : single-wave sequential greedy with __ballot (wave32)
//   update     : atomic segment-sum + convergence gate via device flag
// ---------------------------------------------------------------------------

#define NUM_CLUSTERS 256
#define NPTS 32768
#define DIM 256
#define CAP (NPTS / NUM_CLUSTERS)
#define TOL2 1e-8f   // (1e-4)^2, compare squared Frobenius norm

typedef __bf16 v16bf __attribute__((ext_vector_type(16)));
typedef float  v8f   __attribute__((ext_vector_type(8)));
typedef unsigned int v4u __attribute__((ext_vector_type(4)));
typedef unsigned int v8u __attribute__((ext_vector_type(8)));

// ---------------- init / conversion ----------------------------------------

__global__ __launch_bounds__(256)
void k_init(const float* __restrict__ initcb, float* __restrict__ cb_cur,
            int* __restrict__ done, float* __restrict__ ss) {
    int i = blockIdx.x * 256 + threadIdx.x;
    cb_cur[i] = initcb[i];
    if (i == 0) { *done = 0; ss[0] = 0.f; ss[1] = 0.f; }
}

__device__ __forceinline__ float wave_sum(float s) {
#pragma unroll
    for (int o = 16; o > 0; o >>= 1) s += __shfl_down(s, o, 32);
    return s;
}

__global__ __launch_bounds__(256)
void k_dataconv(const float* __restrict__ data, __bf16* __restrict__ dbf,
                float* __restrict__ dnorm) {
    __shared__ float part[8];
    int n = blockIdx.x, t = threadIdx.x;
    float f = data[(size_t)n * DIM + t];
    dbf[(size_t)n * DIM + t] = (__bf16)f;
    float s = wave_sum(f * f);
    if ((t & 31) == 0) part[t >> 5] = s;
    __syncthreads();
    if (t == 0) {
        float tot = 0.f;
#pragma unroll
        for (int i = 0; i < 8; ++i) tot += part[i];
        dnorm[n] = tot;
    }
}

__global__ __launch_bounds__(256)
void k_cbconv(const float* __restrict__ cb, __bf16* __restrict__ cbf,
              float* __restrict__ cnorm, const int* __restrict__ gate) {
    if (*gate) return;
    __shared__ float part[8];
    int n = blockIdx.x, t = threadIdx.x;
    float f = cb[(size_t)n * DIM + t];
    cbf[(size_t)n * DIM + t] = (__bf16)f;
    float s = wave_sum(f * f);
    if ((t & 31) == 0) part[t >> 5] = s;
    __syncthreads();
    if (t == 0) {
        float tot = 0.f;
#pragma unroll
        for (int i = 0; i < 8; ++i) tot += part[i];
        cnorm[n] = tot;
    }
}

// ---------------- distance GEMM (WMMA bf16, async-to-LDS B tile) -----------
// ISA 7.12.2 16-bit A 16x32 layout: lanes 0-15 hold K bytes [0,16)+[32,48),
// lanes 16-31 hold K bytes [16,32)+[48,64) of each 64B K-chunk of their row.
// B (32x16) uses the same pattern (codebook row-major == B column-major).

__device__ __forceinline__ v16bf load_frag(const __bf16* base, int kc) {
    const v4u* p = (const v4u*)((const char*)base + kc * 64);
    v4u lo = p[0];
    v4u hi = p[2];  // +32 bytes
    v8u w;
    w[0] = lo[0]; w[1] = lo[1]; w[2] = lo[2]; w[3] = lo[3];
    w[4] = hi[0]; w[5] = hi[1]; w[6] = hi[2]; w[7] = hi[3];
    return __builtin_bit_cast(v16bf, w);
}

__global__ __launch_bounds__(256)
void k_gemm_dist(const __bf16* __restrict__ A, const __bf16* __restrict__ B,
                 const float* __restrict__ dn, const float* __restrict__ cn,
                 float* __restrict__ dist, const int* __restrict__ gate) {
    if (*gate) return;
    __shared__ alignas(16) __bf16 ldsB[64 * DIM];   // 32 KB codebook tile
    const int t = threadIdx.x;
    const int colBase = blockIdx.x * 64;

    // --- stage the 64-column codebook tile into LDS with async copies ------
    {
        const char* gsrc = (const char*)(B + (size_t)colBase * DIM);
        unsigned ldsBase = (unsigned)(uintptr_t)(&ldsB[0]);
#pragma unroll
        for (int i = 0; i < 8; ++i) {
            int e = t + i * 256;  // 16-byte unit index, 2048 units total
            unsigned lds = ldsBase + (unsigned)(e * 16);
            unsigned long long g = (unsigned long long)(uintptr_t)(gsrc + (size_t)e * 16);
            asm volatile("global_load_async_to_lds_b128 %0, %1, off"
                         :: "v"(lds), "v"(g) : "memory");
        }
        asm volatile("s_wait_asynccnt 0x0" ::: "memory");
    }
    __syncthreads();

    const int lane = t & 31;
    const int wave = t >> 5;
    const int l15  = lane & 15;
    const int lh   = lane >> 4;
    const int rowBase = blockIdx.y * 128 + wave * 16;

    const __bf16* abase = A + (size_t)(rowBase + l15) * DIM + lh * 8;
    const __bf16* pb0 = &ldsB[( 0 + l15) * DIM + lh * 8];
    const __bf16* pb1 = &ldsB[(16 + l15) * DIM + lh * 8];
    const __bf16* pb2 = &ldsB[(32 + l15) * DIM + lh * 8];
    const __bf16* pb3 = &ldsB[(48 + l15) * DIM + lh * 8];

    v8f acc0 = {}, acc1 = {}, acc2 = {}, acc3 = {};
    v16bf a_cur = load_frag(abase, 0);
#pragma unroll
    for (int kc = 0; kc < 8; ++kc) {
        v16bf b0 = load_frag(pb0, kc);
        v16bf b1 = load_frag(pb1, kc);
        v16bf b2 = load_frag(pb2, kc);
        v16bf b3 = load_frag(pb3, kc);
        v16bf a_nxt = a_cur;
        if (kc < 7) {
            __builtin_prefetch((const char*)abase + (kc + 2) * 64, 0, 3);
            a_nxt = load_frag(abase, kc + 1);   // software-pipelined A fragment
        }
        acc0 = __builtin_amdgcn_wmma_f32_16x16x32_bf16(false, a_cur, false, b0, (short)0, acc0, false, false);
        acc1 = __builtin_amdgcn_wmma_f32_16x16x32_bf16(false, a_cur, false, b1, (short)0, acc1, false, false);
        acc2 = __builtin_amdgcn_wmma_f32_16x16x32_bf16(false, a_cur, false, b2, (short)0, acc2, false, false);
        acc3 = __builtin_amdgcn_wmma_f32_16x16x32_bf16(false, a_cur, false, b3, (short)0, acc3, false, false);
        a_cur = a_nxt;
    }

    // C layout: VGPR r -> lanes0-15: M=r, N=lane ; lanes16-31: M=8+r, N=lane-16
    const int rEoff = lh * 8;
    const int c0 = colBase +  0 + l15;
    const int c1 = colBase + 16 + l15;
    const int c2 = colBase + 32 + l15;
    const int c3 = colBase + 48 + l15;
    const float n0 = cn[c0], n1 = cn[c1], n2 = cn[c2], n3 = cn[c3];
#pragma unroll
    for (int r = 0; r < 8; ++r) {
        int row = rowBase + rEoff + r;
        float d = dn[row];
        size_t base = (size_t)row * NUM_CLUSTERS;
        dist[base + c0] = fmaxf(d + n0 - 2.0f * acc0[r], 0.0f);
        dist[base + c1] = fmaxf(d + n1 - 2.0f * acc1[r], 0.0f);
        dist[base + c2] = fmaxf(d + n2 - 2.0f * acc2[r], 0.0f);
        dist[base + c3] = fmaxf(d + n3 - 2.0f * acc3[r], 0.0f);
    }
}

// ---------------- per-row argsort (bitonic, stable via index in low bits) --

__global__ __launch_bounds__(256)
void k_sort_prefs(const float* __restrict__ dist, unsigned char* __restrict__ prefs,
                  const int* __restrict__ gate) {
    if (*gate) return;
    __shared__ unsigned long long s[256];
    int n = blockIdx.x, t = threadIdx.x;
    float dv = dist[(size_t)n * NUM_CLUSTERS + t];  // >= 0, so uint bits order == value order
    s[t] = ((unsigned long long)__float_as_uint(dv) << 32) | (unsigned)t;
    __syncthreads();
    for (int k = 2; k <= 256; k <<= 1) {
        for (int j = k >> 1; j > 0; j >>= 1) {
            int ixj = t ^ j;
            if (ixj > t) {
                unsigned long long a = s[t], b = s[ixj];
                bool up = ((t & k) == 0);
                if ((a > b) == up) { s[t] = b; s[ixj] = a; }
            }
            __syncthreads();
        }
    }
    prefs[(size_t)n * NUM_CLUSTERS + t] = (unsigned char)(s[t] & 0xFFu);
}

// ---------------- sequential greedy balanced assignment (1 wave) -----------

__global__ __launch_bounds__(32)
void k_greedy(const unsigned char* __restrict__ prefs, int* __restrict__ labels,
              float* __restrict__ labels_f, const int* __restrict__ gate) {
    if (*gate) return;
    __shared__ int cnt[NUM_CLUSTERS];
    int lane = threadIdx.x;
    for (int i = lane; i < NUM_CLUSTERS; i += 32) cnt[i] = 0;
    __syncthreads();
    for (int n = 0; n < NPTS; ++n) {
        int chosen = 0;
#pragma unroll 1
        for (int c = 0; c < NUM_CLUSTERS / 32; ++c) {
            int pref = (int)prefs[(size_t)n * NUM_CLUSTERS + c * 32 + lane];
            bool avail = cnt[pref] < CAP;
            unsigned int m32 = (unsigned int)__ballot(avail);
            if (m32) {
                int first = __ffs(m32) - 1;
                chosen = __shfl(pref, first, 32);
                if (lane == 0) cnt[chosen] = cnt[chosen] + 1;
                break;
            }
        }
        if (lane == 0) { labels[n] = chosen; labels_f[n] = (float)chosen; }
    }
}

// ---------------- codebook update ------------------------------------------

__global__ __launch_bounds__(256)
void k_zero(float* __restrict__ sums, int* __restrict__ cnts) {
    sums[blockIdx.x * 256 + threadIdx.x] = 0.f;
    if (threadIdx.x == 0) cnts[blockIdx.x] = 0;
}

__global__ __launch_bounds__(256)
void k_scatter(const float* __restrict__ data, const int* __restrict__ labels,
               float* __restrict__ sums, int* __restrict__ cnts,
               const int* __restrict__ gate) {
    if (*gate) return;
    int n = blockIdx.x, d = threadIdx.x;
    int k = labels[n];
    atomicAdd(&sums[(size_t)k * DIM + d], data[(size_t)n * DIM + d]);
    if (d == 0) atomicAdd(&cnts[k], 1);
}

__global__ __launch_bounds__(256)
void k_finalize(const float* __restrict__ sums, const int* __restrict__ cnts,
                const float* __restrict__ cb_cur, float* __restrict__ cb_next,
                float* __restrict__ ss, const int* __restrict__ gate) {
    if (*gate) return;
    int k = blockIdx.x, d = threadIdx.x;
    size_t i = (size_t)k * DIM + d;
    int c = cnts[k];
    float oldv = cb_cur[i];
    float newv = (c > 0) ? (sums[i] / (float)c) : oldv;   // empty keeps old centroid
    cb_next[i] = newv;
    float diff = newv - oldv;
    float s = wave_sum(diff * diff);
    if ((threadIdx.x & 31) == 0) atomicAdd(ss, s);
}

__global__ __launch_bounds__(256)
void k_commit(const float* __restrict__ ss, float* __restrict__ cb_cur,
              const float* __restrict__ cb_next, const int* __restrict__ gate) {
    if (*gate) return;
    if (*ss < TOL2) return;  // converged: codebook stays old
    int i = blockIdx.x * 256 + threadIdx.x;
    cb_cur[i] = cb_next[i];
}

__global__ void k_doneset(const float* __restrict__ ss, int* __restrict__ done) {
    if (!*done && *ss < TOL2) *done = 1;
}

// ---------------- host-side orchestration ----------------------------------

extern "C" void kernel_launch(void* const* d_in, const int* in_sizes, int n_in,
                              void* d_out, int out_size, void* d_ws, size_t ws_size,
                              hipStream_t stream) {
    const float* data    = (const float*)d_in[0];   // [32768,256] f32
    const float* init_cb = (const float*)d_in[1];   // [256,256]  f32

    float* out      = (float*)d_out;
    float* cb_cur   = out;                 // codebook output lives here
    float* labels_f = out + NUM_CLUSTERS * DIM;

    // workspace layout (~60 MB total)
    char* w = (char*)d_ws;
    __bf16*        data_bf = (__bf16*)(w);                    // 16,777,216
    __bf16*        cb_bf   = (__bf16*)(w + 16777216);         //    131,072
    float*         dnorm   = (float*) (w + 16908288);         //    131,072
    float*         cnorm   = (float*) (w + 17039360);         //      1,024
    float*         dist    = (float*) (w + 17040384);         // 33,554,432
    unsigned char* prefs   = (unsigned char*)(w + 50594816);  //  8,388,608
    int*           labels  = (int*)   (w + 58983424);         //    131,072
    float*         sums    = (float*) (w + 59114496);         //    262,144
    int*           cnts    = (int*)   (w + 59376640);         //      1,024
    float*         cb_next = (float*) (w + 59377664);         //    262,144
    float*         ss      = (float*) (w + 59639808);         // ss[0], ss[1]
    int*           done    = (int*)   (w + 59639816);

    k_init<<<NUM_CLUSTERS, 256, 0, stream>>>(init_cb, cb_cur, done, ss);
    k_dataconv<<<NPTS, 256, 0, stream>>>(data, data_bf, dnorm);

    for (int it = 0; it < 2; ++it) {
        k_cbconv<<<NUM_CLUSTERS, 256, 0, stream>>>(cb_cur, cb_bf, cnorm, done);
        k_gemm_dist<<<dim3(NUM_CLUSTERS / 64, NPTS / 128), 256, 0, stream>>>(
            data_bf, cb_bf, dnorm, cnorm, dist, done);
        k_sort_prefs<<<NPTS, 256, 0, stream>>>(dist, prefs, done);
        k_greedy<<<1, 32, 0, stream>>>(prefs, labels, labels_f, done);
        k_zero<<<NUM_CLUSTERS, 256, 0, stream>>>(sums, cnts);
        k_scatter<<<NPTS, 256, 0, stream>>>(data, labels, sums, cnts, done);
        k_finalize<<<NUM_CLUSTERS, 256, 0, stream>>>(sums, cnts, cb_cur, cb_next, ss + it, done);
        k_commit<<<NUM_CLUSTERS, 256, 0, stream>>>(ss + it, cb_cur, cb_next, done);
        k_doneset<<<1, 1, 0, stream>>>(ss + it, done);
    }
}
